// ATNLPmodel_51874615001690
// MI455X (gfx1250) — matrix operand order repaired and checked by reading, not compiled, not run on hardware
//
#include <hip/hip_runtime.h>
#include <cstdint>
#include <cmath>

// ---------------- problem constants ----------------
#define BATCH   8
#define L1      2048
#define CHN     128
#define QQ      8
#define L2PIX   64
#define NDB     20000
#define NSHIFT  5
#define MROWS   (NSHIFT * BATCH * QQ)   // 320
#define KDIM    (CHN * L2PIX)           // 8192
#define EPS     1e-8f

#define PANEL_ROWS 128                  // N rows per block panel
#define KSTEP      32                   // K elements per panel
#define LDS_PITCH  36                   // 32 data + 4 pad floats (TDM pad_enable)

// trivial vector types (safe in unions, unlike HIP's float4 class)
typedef __attribute__((ext_vector_type(4)))  float        f32x4;
typedef __attribute__((ext_vector_type(4)))  unsigned int u32x4;
typedef __attribute__((ext_vector_type(4)))  int          i32x4;
typedef __attribute__((ext_vector_type(8)))  int          i32x8;
typedef __attribute__((ext_vector_type(16))) _Float16     v16h;
typedef __attribute__((ext_vector_type(8)))  float        v8f;

// ---------------------------------------------------------------------------
// Kernel 1: build sparse one-hot A matrix (f16 bits), M=320 rows, K=8192.
// m = s*64 + bq ; nonzero at k = ch(l)*64 + ((l + s - 2) mod 64), value 1/8.
// ---------------------------------------------------------------------------
__global__ void build_A_kernel(const int* __restrict__ seq,
                               const int* __restrict__ starts,
                               const int* __restrict__ lens,
                               uint16_t* __restrict__ A) {
    const int bq = blockIdx.x;          // b*Q + q
    const int b  = bq >> 3;
    const int l  = threadIdx.x;         // 0..63

    for (int s = 0; s < NSHIFT; ++s) {
        uint16_t* row = A + (size_t)(s * 64 + bq) * KDIM;
        for (int e = l; e < KDIM; e += 64) row[e] = 0;
    }
    __syncthreads();

    int ln = lens[bq];
    if (ln < 1) ln = 1;
    int pos = starts[bq] + (ln * l) / L2PIX;
    pos = pos < 0 ? 0 : (pos > L1 - 1 ? L1 - 1 : pos);
    const int ch = seq[b * L1 + pos];

    for (int s = 0; s < NSHIFT; ++s) {
        const int pix = (l + s + 62) & 63;     // (l + (s-2)) mod 64
        A[(size_t)(s * 64 + bq) * KDIM + ch * L2PIX + pix] = 0x3000; // 0.125 f16
    }
}

// ---------------------------------------------------------------------------
// Kernel 2: inv_norm[n] = 1 / (||database[n]|| + eps). One wave per row.
// ---------------------------------------------------------------------------
__global__ void norms_kernel(const float* __restrict__ db,
                             float* __restrict__ inv_norm) {
    const int wid  = threadIdx.x >> 5;
    const int lane = threadIdx.x & 31;
    const int row  = blockIdx.x * 8 + wid;
    if (row >= NDB) return;
    const f32x4* p = (const f32x4*)(db + (size_t)row * KDIM);
    float s = 0.f;
    for (int j = lane; j < KDIM / 4; j += 32) {
        f32x4 v = p[j];
        s += v.x * v.x + v.y * v.y + v.z * v.z + v.w * v.w;
    }
    for (int o = 16; o > 0; o >>= 1) s += __shfl_xor(s, o, 32);
    if (lane == 0) inv_norm[row] = 1.f / (sqrtf(s) + EPS);
}

// ---------------------------------------------------------------------------
// Kernel 3: init output: sim <- -inf (atomic-max identity), top_cls <- 0.
// ---------------------------------------------------------------------------
__global__ void init_out_kernel(float* __restrict__ out) {
    const size_t i = (size_t)blockIdx.x * 256 + threadIdx.x;
    const size_t simN = (size_t)BATCH * QQ * NDB;
    if (i < simN) out[i] = -INFINITY;
    if (i < BATCH * QQ) out[simN + i] = 0.f;
}

// ---------------------------------------------------------------------------
// TDM: issue a 2D tensor_load_to_lds of a [PANEL_ROWS x KSTEP] fp32 tile.
// gptr     = global address of tile start (row n0, col k0 of database)
// lds_off  = byte offset of destination buffer in LDS
// rows_rem = NDB - n0 (rows beyond this read back as zeros)
// Padding: 4 DWORDs after every 32 DWORDs -> LDS pitch of 36 floats/row.
// ---------------------------------------------------------------------------
__device__ __forceinline__ void tdm_load_panel(const float* gptr,
                                               uint32_t lds_off,
                                               int rows_rem) {
    const uint64_t ga = (uint64_t)(uintptr_t)gptr;

    union { unsigned int u[4]; u32x4 v; } g0;
    g0.u[0] = 1u;                                     // count=1 (valid), no gather
    g0.u[1] = lds_off;                                // lds_addr
    g0.u[2] = (uint32_t)ga;                           // global_addr[31:0]
    g0.u[3] = (uint32_t)((ga >> 32) & 0x1FFFFFFu)     // global_addr[56:32]
            | (2u << 30);                             // type = 2 (image)

    union { int i[8]; i32x8 v; } g1;
    g1.i[0] = (2 << 16)        // data_size = 2 (4 bytes)
            | (1 << 20)        // pad_enable
            | (4 << 22)        // pad_interval code 4 -> every 32 DWORDs
            | (3 << 25);       // pad_amount  code 3 -> 4 DWORDs
    g1.i[1] = (KDIM & 0xFFFF) << 16;                  // tensor_dim0[15:0]
    g1.i[2] = ((KDIM >> 16) & 0xFFFF)                 // tensor_dim0[31:16]
            | ((rows_rem & 0xFFFF) << 16);            // tensor_dim1[15:0]
    g1.i[3] = ((rows_rem >> 16) & 0xFFFF)             // tensor_dim1[31:16]
            | (KSTEP << 16);                          // tile_dim0 = 32
    g1.i[4] = PANEL_ROWS;                             // tile_dim1 = 128, tile_dim2 = 0
    g1.i[5] = KDIM;                                   // tensor_dim0_stride[31:0]
    g1.i[6] = 0;                                      // stride hi / dim1_stride lo
    g1.i[7] = 0;

    union { int i[4]; i32x4 v; } z4; z4.i[0] = z4.i[1] = z4.i[2] = z4.i[3] = 0;
#if __clang_major__ >= 23
    union { int i[8]; i32x8 v; } z8;
    for (int t = 0; t < 8; ++t) z8.i[t] = 0;
    __builtin_amdgcn_tensor_load_to_lds(g0.v, g1.v, z4.v, z4.v, z8.v, 0);
#else
    __builtin_amdgcn_tensor_load_to_lds(g0.v, g1.v, z4.v, z4.v, 0);
#endif
}

// ---------------------------------------------------------------------------
// Kernel 4: WMMA GEMM with TDM double-buffered DB streaming.
// C[320 x 20000] = A[320 x 8192] * DB^T, f16 in / f32 accumulate.
// Block: 320 threads (10 waves); block owns 128 N-cols; wave owns 32 M-rows.
// Wave 0 drives the Tensor Data Mover; panels land in LDS (pitch 36),
// all waves read fragments via ds_load, convert fp32->f16, and WMMA.
// Epilogue folds max-over-5-shifts with global_atomic_max_num_f32.
// ---------------------------------------------------------------------------
__global__ void __launch_bounds__(320, 1)
gemm_kernel(const uint16_t* __restrict__ A,
            const float* __restrict__ db,
            const float* __restrict__ inv_norm,
            float* __restrict__ out_sim) {
    __shared__ float panel[2][PANEL_ROWS * LDS_PITCH];   // 2 x 18 KB

    const int wave = threadIdx.x >> 5;
    const int lane = threadIdx.x & 31;
    const int half = lane >> 4;         // 0: lanes 0-15, 1: lanes 16-31
    const int lr   = lane & 15;
    const int n0   = blockIdx.x * PANEL_ROWS;
    const int m0   = wave * 32;
    const int rows_rem = NDB - n0;

    const uint32_t lds_base[2] = {
        (uint32_t)(uintptr_t)(&panel[0][0]),
        (uint32_t)(uintptr_t)(&panel[1][0])
    };

    v8f c[2][8];
    const v8f vzero = {0.f, 0.f, 0.f, 0.f, 0.f, 0.f, 0.f, 0.f};
#pragma unroll
    for (int i = 0; i < 2; ++i)
#pragma unroll
        for (int j = 0; j < 8; ++j) c[i][j] = vzero;

    // prologue: DMA first panel
    if (wave == 0)
        tdm_load_panel(db + (size_t)n0 * KDIM, lds_base[0], rows_rem);

    for (int k0 = 0, it = 0; k0 < KDIM; k0 += KSTEP, ++it) {
        const int cur = it & 1;

        if (wave == 0) __builtin_amdgcn_s_wait_tensorcnt(0);
        __syncthreads();   // panel[cur] ready; everyone done reading panel[cur^1]

        if (wave == 0 && k0 + KSTEP < KDIM)
            tdm_load_panel(db + (size_t)n0 * KDIM + (k0 + KSTEP),
                           lds_base[cur ^ 1], rows_rem);

        // ---- B fragments from LDS (32x16 f16): col=lr, K = k0+half*16 .. +15
        v16h bfrag[8];
#pragma unroll
        for (int j = 0; j < 8; ++j) {
            const int prow = j * 16 + lr;
            const float* lp = &panel[cur][prow * LDS_PITCH + half * 16];
            union { f32x4 q[4]; float f[16]; } st;
            const f32x4* p4 = (const f32x4*)lp;
            st.q[0] = p4[0]; st.q[1] = p4[1]; st.q[2] = p4[2]; st.q[3] = p4[3];
            v16h hv;
#pragma unroll
            for (int e = 0; e < 16; ++e) hv[e] = (_Float16)st.f[e];
            bfrag[j] = hv;
        }

        // ---- per M-tile: A fragment (16x32 f16) from global, 8 WMMAs
#pragma unroll
        for (int i = 0; i < 2; ++i) {
            union { u32x4 q[2]; v16h v; } af;
            const u32x4* ap =
                (const u32x4*)(A + (size_t)(m0 + i * 16 + lr) * KDIM + k0 + half * 16);
            af.q[0] = ap[0];   // K = k0 + half*16 .. +7
            af.q[1] = ap[1];   // K = k0 + half*16 + 8 .. +15
#pragma unroll
            for (int j = 0; j < 8; ++j)
                c[i][j] = __builtin_amdgcn_wmma_f32_16x16x32_f16(
                    false, af.v, false, bfrag[j], (short)0, c[i][j], false, false);
        }
    }

    // ---- epilogue: scale by inv_norm, fold shift-max with native f32 atomic max
#pragma unroll
    for (int i = 0; i < 2; ++i) {
#pragma unroll
        for (int j = 0; j < 8; ++j) {
            const int n = n0 + j * 16 + lr;
            if (n < NDB) {
                const float sc = inv_norm[n];
#pragma unroll
                for (int r = 0; r < 8; ++r) {
                    const int m  = m0 + i * 16 + r + half * 8;
                    const int bq = m & 63;                  // m = s*64 + bq
                    float v = c[i][j][r] * sc;
                    float* addr = out_sim + (size_t)bq * NDB + n;
                    asm volatile("global_atomic_max_num_f32 %0, %1, off"
                                 :: "v"(addr), "v"(v) : "memory");
                }
            }
        }
    }
}

// ---------------------------------------------------------------------------
// Kernel 5: per-(b,q) argmax over N, class lookup, store as float.
// ---------------------------------------------------------------------------
__global__ void argmax_kernel(const float* __restrict__ sim,
                              const int* __restrict__ classes,
                              float* __restrict__ top) {
    __shared__ float sv[256];
    __shared__ int   si[256];
    const int bq = blockIdx.x;
    const float* row = sim + (size_t)bq * NDB;
    float best = -INFINITY;
    int   bi   = 0;
    for (int n = threadIdx.x; n < NDB; n += 256) {
        float v = row[n];
        if (v > best || (v == best && n < bi)) { best = v; bi = n; }
    }
    sv[threadIdx.x] = best;
    si[threadIdx.x] = bi;
    __syncthreads();
    for (int o = 128; o > 0; o >>= 1) {
        if (threadIdx.x < o) {
            const float vo = sv[threadIdx.x + o];
            const int   io = si[threadIdx.x + o];
            if (vo > sv[threadIdx.x] ||
                (vo == sv[threadIdx.x] && io < si[threadIdx.x])) {
                sv[threadIdx.x] = vo;
                si[threadIdx.x] = io;
            }
        }
        __syncthreads();
    }
    if (threadIdx.x == 0) top[bq] = (float)classes[si[0]];
}

// ---------------------------------------------------------------------------
extern "C" void kernel_launch(void* const* d_in, const int* in_sizes, int n_in,
                              void* d_out, int out_size, void* d_ws, size_t ws_size,
                              hipStream_t stream) {
    const int*   seq     = (const int*)d_in[0];   // (B, L1)
    const int*   starts  = (const int*)d_in[1];   // (B, Q)
    const int*   lens    = (const int*)d_in[2];   // (B, Q)
    const float* db      = (const float*)d_in[3]; // (N, C, L2)
    const int*   classes = (const int*)d_in[4];   // (N,)
    float* out = (float*)d_out;                   // sim (B,Q,N) ++ top_cls (B,Q)

    uint16_t* A        = (uint16_t*)d_ws;                                   // 5,242,880 B
    float*    inv_norm = (float*)((char*)d_ws + (size_t)MROWS * KDIM * 2);  // 80,000 B

    build_A_kernel<<<BATCH * QQ, L2PIX, 0, stream>>>(seq, starts, lens, A);
    norms_kernel<<<NDB / 8, 256, 0, stream>>>(db, inv_norm);

    const size_t outN = (size_t)BATCH * QQ * NDB + BATCH * QQ;
    init_out_kernel<<<(unsigned)((outN + 255) / 256), 256, 0, stream>>>(out);

    gemm_kernel<<<(NDB + PANEL_ROWS - 1) / PANEL_ROWS, 320, 0, stream>>>(
        A, db, inv_norm, out);

    argmax_kernel<<<BATCH * QQ, 256, 0, stream>>>(out, classes,
                                                  out + (size_t)BATCH * QQ * NDB);
}
